// HyperbolicTransformer_35321811042421
// MI455X (gfx1250) — matrix operand order repaired and checked by reading, not compiled, not run on hardware
//
#include <hip/hip_runtime.h>
#include <stdint.h>

#define MINN     1e-15f
#define MAXNORM  0.996f          // (1 - 4e-3)/sqrt(c), c=1
#define NEGS     0.2f
#define D_DST    20000
#define N_EDGE   320000

typedef __attribute__((ext_vector_type(16))) __bf16 bf16x16;
typedef __attribute__((ext_vector_type(8)))  float  f32x8;

union FragU {
  uint32_t u[8];
  uint4    q[2];
  bf16x16  v;
};

// ---- workspace layout (float/dword offsets) ----
static constexpr size_t OFF_WF   = 0;                                  // 156*256 packed bf16 B-frags
static constexpr size_t OFF_HB   = 39936;                              // 200: hb_k | hb_v
static constexpr size_t OFF_HBQ  = 40136;                              // 100
static constexpr size_t OFF_HBO  = 40236;                              // 100
static constexpr size_t OFF_ZTF  = 40336;                              // 100
static constexpr size_t OFF_N2   = 40436;                              // 8 scalars
static constexpr size_t OFF_WQT  = 40448;                              // 20000 (Wq transposed)
static constexpr size_t OFF_WOT  = 60448;                              // 20000 (Wo transposed)
static constexpr size_t OFF_Q    = 80448;                              // D*100 f32
static constexpr size_t OFF_DSTH = OFF_Q    + (size_t)D_DST*100;       // D*100 f32
static constexpr size_t OFF_V    = OFF_DSTH + (size_t)D_DST*100;       // E*100 f16 (= E*50 dwords)
static constexpr size_t OFF_SC   = OFF_V    + (size_t)N_EDGE*50;       // E*2 f32
static constexpr size_t OFF_SMAX = OFF_SC   + (size_t)N_EDGE*2;        // D*2 (uint keys)
static constexpr size_t OFF_SSUM = OFF_SMAX + (size_t)D_DST*2;         // D*2 f32
static constexpr size_t OFF_HAGG = OFF_SSUM + (size_t)D_DST*2;         // D*100 f32
static constexpr size_t ZERO_DW  = (size_t)D_DST*2 + (size_t)D_DST*2 + (size_t)D_DST*100;

// ---- helpers ----
__device__ __forceinline__ float wred32(float v) {
#pragma unroll
  for (int off = 16; off > 0; off >>= 1) v += __shfl_xor(v, off, 32);
  return v;
}

__device__ __forceinline__ float artanh_(float x) {
  x = fminf(fmaxf(x, -1.0f + 1e-7f), 1.0f - 1e-7f);
  return 0.5f * (log1pf(x) - log1pf(-x));
}

__device__ __forceinline__ unsigned fkey(float x) {
  unsigned u = __float_as_uint(x);
  return (u & 0x80000000u) ? ~u : (u | 0x80000000u);
}
__device__ __forceinline__ float kinv(unsigned k) {
  unsigned u = (k & 0x80000000u) ? (k & 0x7FFFFFFFu) : ~k;
  return __uint_as_float(u);
}

// hyp_linear epilogue on a 100-length row held as v[k] at idx=lane+32k
__device__ __forceinline__ void hyp_post(float* v, int lane, float xn,
                                         const float* __restrict__ hb, float y2) {
  float ss = 0.f;
#pragma unroll
  for (int k = 0; k < 4; ++k) { int idx = lane + 32*k; if (idx < 100) ss += v[k]*v[k]; }
  float mxn2 = wred32(ss);
  float mxn  = fmaxf(sqrtf(mxn2), MINN);
  float tt   = fminf(tanhf((mxn / xn) * artanh_(xn)), MAXNORM);
  float s1   = tt / mxn;
  float x2   = tt * tt;
  float xyp  = 0.f;
#pragma unroll
  for (int k = 0; k < 4; ++k) { int idx = lane + 32*k; v[k] *= s1; if (idx < 100) xyp += v[k]*hb[idx]; }
  float xy  = wred32(xyp);
  float den = fmaxf(1.f + 2.f*xy + x2*y2, MINN);
  float A   = (1.f + 2.f*xy + y2) / den;
  float B   = (1.f - x2) / den;
  float ss2 = 0.f;
#pragma unroll
  for (int k = 0; k < 4; ++k) {
    int idx = lane + 32*k;
    float o = (idx < 100) ? (A*v[k] + B*hb[idx]) : 0.f;
    v[k] = o; ss2 += o*o;
  }
  float n  = fmaxf(sqrtf(wred32(ss2)), MINN);
  float s2 = (n > MAXNORM) ? MAXNORM / n : 1.f;
#pragma unroll
  for (int k = 0; k < 4; ++k) v[k] *= s2;
}

// encode one feature segment: hyp_encode(src[0:len]) -> bf16 dst; returns ||encoded||^2
__device__ __forceinline__ float enc_seg(const float* __restrict__ src, int len,
                                         __bf16* __restrict__ dst, int lane) {
  float v[6]; float ss = 0.f;
#pragma unroll
  for (int k = 0; k < 6; ++k) {
    int idx = lane + 32*k;
    float x = (idx < len) ? src[idx] : 0.f;
    v[k] = x; ss += x*x;
  }
  float n2 = wred32(ss);
  float n  = fmaxf(sqrtf(n2), MINN);
  float s  = fminf(tanhf(n), MAXNORM) / n;
#pragma unroll
  for (int k = 0; k < 6; ++k) {
    int idx = lane + 32*k;
    if (idx < len) dst[idx] = (__bf16)(v[k] * s);
  }
  return s*s*n2;
}

__device__ __forceinline__ float enc_time(float dte, const float* __restrict__ tw,
                                          const float* __restrict__ tb,
                                          __bf16* __restrict__ dst, int lane) {
  float v[4]; float ss = 0.f;
#pragma unroll
  for (int k = 0; k < 4; ++k) {
    int idx = lane + 32*k;
    float x = (idx < 100) ? cosf(dte * tw[idx] + tb[idx]) : 0.f;
    v[k] = x; ss += x*x;
  }
  float n2 = wred32(ss);
  float n  = fmaxf(sqrtf(n2), MINN);
  float s  = fminf(tanhf(n), MAXNORM) / n;
#pragma unroll
  for (int k = 0; k < 4; ++k) {
    int idx = lane + 32*k;
    if (idx < 100) dst[idx] = (__bf16)(v[k] * s);
  }
  return s*s*n2;
}

// ---- prep: pack fused [Wk;Wv] (200x372) into WMMA B fragments, bf16 ----
// frag layout: [ct(13)][kc(12)][lane(32)][dword j(8)], lane: col = ct*16+lane%16,
// K = kc*32 + (j/4)*16 + (lane/16)*8 + 2*(j%4) (+1 in high half of dword)
__global__ __launch_bounds__(256) void k_pack(const float* __restrict__ Wk,
                                              const float* __restrict__ Wv,
                                              uint32_t* __restrict__ wf) {
  const int frag = blockIdx.x;            // 0..155
  const int ct = frag / 12, kc = frag % 12;
  const int lane = threadIdx.x >> 3;
  const int j    = threadIdx.x & 7;
  const int col  = ct*16 + (lane & 15);
  const int k0   = kc*32 + (j >> 2)*16 + (lane >> 4)*8 + 2*(j & 3);
  float f0 = 0.f, f1 = 0.f;
  if (col < 200) {
    const float* wr = (col < 100) ? (Wk + (size_t)col*372) : (Wv + (size_t)(col-100)*372);
    if (k0     < 372) f0 = wr[k0];
    if (k0 + 1 < 372) f1 = wr[k0 + 1];
  }
  union { __bf16 h; unsigned short u; } c0, c1;
  c0.h = (__bf16)f0; c1.h = (__bf16)f1;
  wf[(size_t)frag*256 + lane*8 + j] = (uint32_t)c0.u | ((uint32_t)c1.u << 16);
}

// ---- prep: hb vectors, ztf, transposes ----
__global__ __launch_bounds__(256) void k_prep2(const float* __restrict__ bk, const float* __restrict__ bv,
                                               const float* __restrict__ bq, const float* __restrict__ bo,
                                               const float* __restrict__ time_b,
                                               const float* __restrict__ Wq, const float* __restrict__ Wo,
                                               float* __restrict__ ws) {
  const int lane = threadIdx.x & 31, wave = threadIdx.x >> 5;
  float* hbbase = ws + OFF_HB;   // hb_k, hb_v, hb_q, hb_o, ztf each 100, contiguous
  float* n2a    = ws + OFF_N2;
  if (wave < 5) {
    const float* src = (wave == 0) ? bk : (wave == 1) ? bv : (wave == 2) ? bq : bo;
    float v[4]; float ss = 0.f;
#pragma unroll
    for (int k = 0; k < 4; ++k) {
      int idx = lane + 32*k; float x = 0.f;
      if (idx < 100) x = (wave == 4) ? cosf(time_b[idx]) : src[idx];
      v[k] = x; ss += x*x;
    }
    float n2 = wred32(ss);
    float n  = fmaxf(sqrtf(n2), MINN);
    float s  = fminf(tanhf(n), MAXNORM) / n;
#pragma unroll
    for (int k = 0; k < 4; ++k) { int idx = lane + 32*k; if (idx < 100) hbbase[wave*100 + idx] = v[k]*s; }
    if (lane == 0) n2a[wave] = s*s*n2;
  }
  __syncthreads();
  float* WqT = ws + OFF_WQT; float* WoT = ws + OFF_WOT;
  for (int i = threadIdx.x; i < 20000; i += 256) {
    int o = i / 200, j = i % 200;
    WqT[j*100 + o] = Wq[i];
    WoT[j*100 + o] = Wo[i];
  }
}

__global__ void k_zero(float* __restrict__ p, int n) {
  int i = blockIdx.x * 256 + threadIdx.x;
  if (i < n) p[i] = 0.f;
}

// ---- Q projection per dst node (wave-per-row, f32) ----
__global__ __launch_bounds__(256) void k_q(const float* __restrict__ dst_h, float* __restrict__ ws) {
  __shared__ float xin[8][200];
  const int lane = threadIdx.x & 31;
  const int w    = threadIdx.x >> 5;
  const int d    = blockIdx.x * 8 + w;
  const float* ztf = ws + OFF_ZTF;
  const float* n2a = ws + OFF_N2;
  const float* WqT = ws + OFF_WQT;
  const float* hbq = ws + OFF_HBQ;
  float* Qb = ws + OFF_Q;
  float* Dh = ws + OFF_DSTH;

  float v[4]; float ss = 0.f;
#pragma unroll
  for (int k = 0; k < 4; ++k) {
    int idx = lane + 32*k;
    float x = (idx < 100) ? dst_h[(size_t)d*100 + idx] : 0.f;
    v[k] = x; ss += x*x;
  }
  float n2 = wred32(ss);
  float n  = fmaxf(sqrtf(n2), MINN);
  float s  = fminf(tanhf(n), MAXNORM) / n;
#pragma unroll
  for (int k = 0; k < 4; ++k) {
    int idx = lane + 32*k;
    if (idx < 100) { float e = v[k]*s; xin[w][idx] = e; Dh[(size_t)d*100 + idx] = e; xin[w][100+idx] = ztf[idx]; }
  }
  float xn = fmaxf(sqrtf(s*s*n2 + n2a[4]), MINN);

  float out[4] = {0.f, 0.f, 0.f, 0.f};
  for (int j = 0; j < 200; ++j) {
    float xj = xin[w][j];
    const float* wt = WqT + j*100;
#pragma unroll
    for (int k = 0; k < 4; ++k) { int o = lane + 32*k; if (o < 100) out[k] = fmaf(xj, wt[o], out[k]); }
  }
  hyp_post(out, lane, xn, hbq, n2a[2]);
#pragma unroll
  for (int k = 0; k < 4; ++k) { int idx = lane + 32*k; if (idx < 100) Qb[(size_t)d*100 + idx] = out[k]; }
}

// ---- heavy kernel: 32 edges/block, K/V projection via bf16 WMMA (2 row-tiles share
//      each B fragment), hyperbolic epilogue, attention scores + segment max ----
__global__ __launch_bounds__(128) void k_kv(const float* __restrict__ neigh_h,
                                            const float* __restrict__ edge_f,
                                            const float* __restrict__ dt,
                                            const int*   __restrict__ edge_dst,
                                            const float* __restrict__ time_w,
                                            const float* __restrict__ time_b,
                                            float* __restrict__ ws) {
  __shared__ __bf16 kvA[32][384];    // 32 edges x (372->384) bf16
  __shared__ float  mxs[32][208];    // matmul result: cols 0..99 K, 100..199 V
  __shared__ float  xrow[32];

  const int lane = threadIdx.x & 31;
  const int wave = threadIdx.x >> 5;
  const int base = blockIdx.x * 32;
  const uint32_t* __restrict__ wf = (const uint32_t*)(ws + OFF_WF);
  const float* hbkv = ws + OFF_HB;
  const float* n2a  = ws + OFF_N2;
  const float* Qb   = ws + OFF_Q;
  _Float16* Vb = (_Float16*)(ws + OFF_V);
  float* sc = ws + OFF_SC;
  unsigned* smax = (unsigned*)(ws + OFF_SMAX);

  // phase 1: encode kv_in rows -> LDS bf16, row norms; prefetch Q rows for phase 4
  for (int r = wave*8; r < wave*8 + 8; ++r) {
    const int e = base + r;
    const int dq = edge_dst[e];
    if (lane < 4) __builtin_prefetch((const void*)(Qb + (size_t)dq*100 + lane*32), 0, 1);
    float t1 = enc_seg(neigh_h + (size_t)e*100, 100, &kvA[r][0],   lane);
    float t2 = enc_seg(edge_f  + (size_t)e*172, 172, &kvA[r][100], lane);
    float t3 = enc_time(dt[e], time_w, time_b,       &kvA[r][272], lane);
    if (lane < 12) kvA[r][372 + lane] = (__bf16)0.f;
    if (lane == 0) xrow[r] = fmaxf(sqrtf(t1 + t2 + t3), MINN);
  }
  __syncthreads();

  // phase 2: 13 column tiles, K=384 = 12 chunks of 32; two A row-tiles per B fragment
  const uint32_t* kv32 = (const uint32_t*)&kvA[0][0];
  const int arow  = lane & 15;
  const int ahalf = lane >> 4;
  for (int ct = wave; ct < 13; ct += 4) {
    f32x8 acc0 = {0.f,0.f,0.f,0.f,0.f,0.f,0.f,0.f};
    f32x8 acc1 = {0.f,0.f,0.f,0.f,0.f,0.f,0.f,0.f};
#pragma unroll
    for (int kc = 0; kc < 12; ++kc) {
      FragU a0, a1, b;
      const uint32_t* ap0 = kv32 + arow*192 + kc*16 + ahalf*4;
      const uint32_t* ap1 = ap0 + 16*192;
      a0.q[0] = ((const uint4*)ap0)[0];   // VGPR 0..3: K = kc*32 + half*8 + {0..7}
      a0.q[1] = ((const uint4*)ap0)[2];   // VGPR 4..7: K = kc*32 + 16 + half*8 + {0..7}
      a1.q[0] = ((const uint4*)ap1)[0];
      a1.q[1] = ((const uint4*)ap1)[2];
      const uint4* bp = (const uint4*)(wf + (size_t)(ct*12 + kc)*256 + lane*8);
      b.q[0] = bp[0];
      b.q[1] = bp[1];
      acc0 = __builtin_amdgcn_wmma_f32_16x16x32_bf16(false, a0.v, false, b.v, (short)0, acc0, false, false);
      acc1 = __builtin_amdgcn_wmma_f32_16x16x32_bf16(false, a1.v, false, b.v, (short)0, acc1, false, false);
    }
    const int col   = ct*16 + (lane & 15);
    const int mrow0 = (lane >> 4) * 8;
#pragma unroll
    for (int rr = 0; rr < 8; ++rr) {
      mxs[mrow0 + rr][col]      = acc0[rr];
      mxs[16 + mrow0 + rr][col] = acc1[rr];
    }
  }
  __syncthreads();

  // phase 3: hyperbolic epilogue per half-row (K back into LDS, V -> global f16)
  for (int t = wave; t < 64; t += 4) {
    const int row = t >> 1, part = t & 1;
    const int e = base + row;
    float v[4]; float ss = 0.f;
#pragma unroll
    for (int k = 0; k < 4; ++k) {
      int idx = lane + 32*k;
      float x = (idx < 100) ? mxs[row][part*100 + idx] : 0.f;
      v[k] = x; ss += x*x;
    }
    float mxn2 = wred32(ss);
    float mxn  = fmaxf(sqrtf(mxn2), MINN);
    float xn   = xrow[row];
    float tt   = fminf(tanhf((mxn / xn) * artanh_(xn)), MAXNORM);
    float s1   = tt / mxn;
    float x2   = tt * tt;
    const float* hb = hbkv + part*100;
    float y2 = n2a[part];
    float xyp = 0.f;
#pragma unroll
    for (int k = 0; k < 4; ++k) { int idx = lane + 32*k; v[k] *= s1; if (idx < 100) xyp += v[k]*hb[idx]; }
    float xy  = wred32(xyp);
    float den = fmaxf(1.f + 2.f*xy + x2*y2, MINN);
    float A   = (1.f + 2.f*xy + y2) / den;
    float B   = (1.f - x2) / den;
    float ss2 = 0.f;
#pragma unroll
    for (int k = 0; k < 4; ++k) {
      int idx = lane + 32*k;
      float o = (idx < 100) ? (A*v[k] + B*hb[idx]) : 0.f;
      v[k] = o; ss2 += o*o;
    }
    float nn = fmaxf(sqrtf(wred32(ss2)), MINN);
    float s2 = (nn > MAXNORM) ? MAXNORM / nn : 1.f;
    if (part == 0) {
#pragma unroll
      for (int k = 0; k < 4; ++k) { int idx = lane + 32*k; if (idx < 100) mxs[row][idx] = v[k]*s2; }
    } else {
      _Float16* Vg = Vb + (size_t)e*100;
#pragma unroll
      for (int k = 0; k < 4; ++k) { int idx = lane + 32*k; if (idx < 100) Vg[idx] = (_Float16)(v[k]*s2); }
    }
  }
  __syncthreads();

  // phase 4: per-head scores = leaky_relu(Q . K) + segment max (float-keyed atomicMax)
  for (int r = wave; r < 32; r += 4) {
    const int e = base + r;
    const int d = edge_dst[e];
    const float* Qr = Qb + (size_t)d*100;
    float s0 = 0.f, s1h = 0.f;
#pragma unroll
    for (int k = 0; k < 4; ++k) {
      int idx = lane + 32*k;
      if (idx < 100) {
        float p = Qr[idx] * mxs[r][idx];
        if (idx < 50) s0 += p; else s1h += p;
      }
    }
    s0  = wred32(s0);
    s1h = wred32(s1h);
    if (lane == 0) {
      float a0 = (s0  > 0.f) ? s0  : NEGS * s0;
      float a1 = (s1h > 0.f) ? s1h : NEGS * s1h;
      sc[(size_t)e*2 + 0] = a0;
      sc[(size_t)e*2 + 1] = a1;
      atomicMax(&smax[d*2 + 0], fkey(a0));
      atomicMax(&smax[d*2 + 1], fkey(a1));
    }
  }
}

// ---- softmax pass 1: segment sum of exp(score - smax) ----
__global__ __launch_bounds__(256) void k_soft1(const int* __restrict__ edge_dst, float* __restrict__ ws) {
  int i = blockIdx.x * 256 + threadIdx.x;
  if (i >= N_EDGE * 2) return;
  int e = i >> 1, h = i & 1;
  int d = edge_dst[e];
  const float* scv = ws + OFF_SC;
  const unsigned* smax = (const unsigned*)(ws + OFF_SMAX);
  float* ssum = ws + OFF_SSUM;
  float m = kinv(smax[d*2 + h]);
  float z = expf(scv[i] - m);
  atomicAdd(&ssum[d*2 + h], z);
}

// ---- aggregation: h_agg += att * V (wave per edge; recompute exp deterministically) ----
__global__ __launch_bounds__(256) void k_agg(const int* __restrict__ edge_dst, float* __restrict__ ws) {
  const int lane = threadIdx.x & 31;
  const int wave = threadIdx.x >> 5;
  const int e = blockIdx.x * 8 + wave;
  const int d = edge_dst[e];
  const float* scv  = ws + OFF_SC;
  const float* ssum = ws + OFF_SSUM;
  const unsigned* smax = (const unsigned*)(ws + OFF_SMAX);
  const _Float16* V = (const _Float16*)(ws + OFF_V) + (size_t)e*100;
  float* hagg = ws + OFF_HAGG + (size_t)d*100;
  float att01[2];
#pragma unroll
  for (int h = 0; h < 2; ++h) {
    float m = kinv(smax[d*2 + h]);
    float z = expf(scv[(size_t)e*2 + h] - m);
    att01[h] = z / fmaxf(ssum[d*2 + h], MINN);
  }
#pragma unroll
  for (int k = 0; k < 4; ++k) {
    int idx = lane + 32*k;
    if (idx < 100) {
      float att = (idx >= 50) ? att01[1] : att01[0];
      atomicAdd(&hagg[idx], (float)V[idx] * att);
    }
  }
}

// ---- output: hyp_linear(Wo) + HypAct + logmap0 + layernorm ----
__global__ __launch_bounds__(256) void k_out(const float* __restrict__ ln_g, const float* __restrict__ ln_b,
                                             float* __restrict__ ws, float* __restrict__ outg) {
  __shared__ float xin[8][200];
  const int lane = threadIdx.x & 31;
  const int w    = threadIdx.x >> 5;
  const int d    = blockIdx.x * 8 + w;
  const float* hagg = ws + OFF_HAGG;
  const float* Dh   = ws + OFF_DSTH;
  const float* WoT  = ws + OFF_WOT;
  const float* hbo  = ws + OFF_HBO;
  const float* n2a  = ws + OFF_N2;

  float ssx = 0.f;
#pragma unroll
  for (int k = 0; k < 4; ++k) {
    int idx = lane + 32*k;
    if (idx < 100) {
      float a = hagg[(size_t)d*100 + idx];
      float b = Dh[(size_t)d*100 + idx];
      xin[w][idx] = a; xin[w][100 + idx] = b;
      ssx += a*a + b*b;
    }
  }
  float xn = fmaxf(sqrtf(wred32(ssx)), MINN);

  float out[4] = {0.f, 0.f, 0.f, 0.f};
  for (int j = 0; j < 200; ++j) {
    float xj = xin[w][j];
    const float* wt = WoT + j*100;
#pragma unroll
    for (int k = 0; k < 4; ++k) { int o = lane + 32*k; if (o < 100) out[k] = fmaf(xj, wt[o], out[k]); }
  }
  hyp_post(out, lane, xn, hbo, n2a[3]);

  // HypAct: expmap0(leaky_relu(logmap0(rst)))
  float nr2 = 0.f;
#pragma unroll
  for (int k = 0; k < 4; ++k) { int idx = lane + 32*k; if (idx < 100) nr2 += out[k]*out[k]; }
  float nr = fmaxf(sqrtf(wred32(nr2)), MINN);
  float sl = artanh_(nr) / nr;
  float na2 = 0.f;
#pragma unroll
  for (int k = 0; k < 4; ++k) {
    int idx = lane + 32*k;
    float u = out[k] * sl;
    u = (u > 0.f) ? u : NEGS * u;
    out[k] = u;
    if (idx < 100) na2 += u*u;
  }
  float na = fmaxf(sqrtf(wred32(na2)), MINN);
  float t  = tanhf(na);
  float se = fminf(t, MAXNORM) / na;                  // expmap0 + proj combined
  float nb = fmaxf(fminf(t, MAXNORM), MINN);          // ||rst2||
  float sh = artanh_(nb) / nb;                        // logmap0 scale
  float comb = se * sh;
  float mup = 0.f;
#pragma unroll
  for (int k = 0; k < 4; ++k) { int idx = lane + 32*k; out[k] *= comb; if (idx < 100) mup += out[k]; }
  float mu = wred32(mup) * 0.01f;
  float varp = 0.f;
#pragma unroll
  for (int k = 0; k < 4; ++k) { int idx = lane + 32*k; if (idx < 100) { float dlt = out[k] - mu; varp += dlt*dlt; } }
  float var = wred32(varp) * 0.01f;
  float inv = rsqrtf(var + 1e-5f);
#pragma unroll
  for (int k = 0; k < 4; ++k) {
    int idx = lane + 32*k;
    if (idx < 100) outg[(size_t)d*100 + idx] = (out[k] - mu) * inv * ln_g[idx] + ln_b[idx];
  }
}

extern "C" void kernel_launch(void* const* d_in, const int* in_sizes, int n_in,
                              void* d_out, int out_size, void* d_ws, size_t ws_size,
                              hipStream_t stream) {
  const float* dst_h    = (const float*)d_in[0];
  const float* neigh_h  = (const float*)d_in[1];
  const float* edge_f   = (const float*)d_in[2];
  const float* dt       = (const float*)d_in[3];
  const int*   edge_dst = (const int*)d_in[4];
  const float* Wq       = (const float*)d_in[5];
  const float* bq       = (const float*)d_in[6];
  const float* Wk       = (const float*)d_in[7];
  const float* bk       = (const float*)d_in[8];
  const float* Wv       = (const float*)d_in[9];
  const float* bv       = (const float*)d_in[10];
  const float* Wo       = (const float*)d_in[11];
  const float* bo       = (const float*)d_in[12];
  const float* time_w   = (const float*)d_in[13];
  const float* time_b   = (const float*)d_in[14];
  const float* ln_g     = (const float*)d_in[15];
  const float* ln_b     = (const float*)d_in[16];
  float* out = (float*)d_out;
  float* ws  = (float*)d_ws;

  // prep
  k_pack<<<156, 256, 0, stream>>>(Wk, Wv, (uint32_t*)(ws + OFF_WF));
  k_prep2<<<1, 256, 0, stream>>>(bk, bv, bq, bo, time_b, Wq, Wo, ws);
  k_zero<<<(int)((ZERO_DW + 255) / 256), 256, 0, stream>>>(ws + OFF_SMAX, (int)ZERO_DW);
  // Q per dst node
  k_q<<<D_DST / 8, 256, 0, stream>>>(dst_h, ws);
  // K/V + scores (WMMA), 32 edges per block
  k_kv<<<N_EDGE / 32, 128, 0, stream>>>(neigh_h, edge_f, dt, edge_dst, time_w, time_b, ws);
  // edge softmax
  k_soft1<<<(N_EDGE * 2) / 256, 256, 0, stream>>>(edge_dst, ws);
  k_agg<<<N_EDGE / 8, 256, 0, stream>>>(edge_dst, ws);
  // output head
  k_out<<<D_DST / 8, 256, 0, stream>>>(ln_g, ln_b, ws, out);
}
